// MultiHeadAttention_31233002176572
// MI455X (gfx1250) — matrix operand order repaired and checked by reading, not compiled, not run on hardware
//
#include <hip/hip_runtime.h>

typedef __bf16 bf16_t;
typedef __attribute__((ext_vector_type(4)))  __bf16 bf16x4;
typedef __attribute__((ext_vector_type(8)))  __bf16 bf16x8;
typedef __attribute__((ext_vector_type(16))) __bf16 v16bf;
typedef __attribute__((ext_vector_type(4)))  float  f32x4;
typedef __attribute__((ext_vector_type(8)))  float  v8f;

union FragA { v16bf v; bf16x8 h[2]; };

__device__ __forceinline__ bf16_t f2bf(float f) {
  union { float f; unsigned u; } in; in.f = f;
  unsigned u = in.u;
  u += 0x7FFFu + ((u >> 16) & 1u);           // round-to-nearest-even
  union { unsigned short s; bf16_t b; } out;
  out.s = (unsigned short)(u >> 16);
  return out.b;
}

__device__ __forceinline__ v8f wmma_bf16(v16bf a, v16bf b, v8f c) {
  return __builtin_amdgcn_wmma_f32_16x16x32_bf16(
      /*neg_a=*/false, a, /*neg_b=*/false, b,
      /*c_mod=*/(short)0, c, /*reuse_a=*/false, /*reuse_b=*/false);
}

// ---------------------------------------------------------------- convert
// n4 = element count / 4 (all tensors are multiples of 1024)
__global__ __launch_bounds__(256)
void cvt_k(const float* __restrict__ src, bf16_t* __restrict__ dst, int n4) {
  int i = blockIdx.x * 256 + threadIdx.x;
  const int stride = gridDim.x * 256;
  for (; i < n4; i += stride) {
    f32x4 v = ((const f32x4*)src)[i];
    bf16x4 o;
    #pragma unroll
    for (int j = 0; j < 4; j++) o[j] = f2bf(v[j]);
    ((bf16x4*)dst)[i] = o;
  }
}

// ---------------------------------------------------------------- GEMM
// C[M][N] = A[M][K](bf16) @ Bw[N][K]^T(bf16).  Bw row-major [N][K] -> rows are
// k-contiguous, matching the WMMA B-operand fragment directly.
// 128x128 tile, K-step 64, software-pipelined global->reg->LDS staging.
template<bool WRITE_F32, bool QKV_SCALE>
__global__ __launch_bounds__(256)
void gemm_bf16_k(const bf16_t* __restrict__ A, const bf16_t* __restrict__ Bw,
                 bf16_t* __restrict__ Cb, float* __restrict__ Cf,
                 const float* __restrict__ bias,
                 int M, int N, int K, float scale, int scaleCols)
{
  __shared__ bf16_t As[128][72];   // [m][k] stride 144B: 16B aligned, conflict-spread
  __shared__ bf16_t Bs[128][72];   // [n][k]

  const int t    = threadIdx.x;
  const int lane = t & 31, wid = t >> 5;
  const int m0 = blockIdx.y * 128, n0 = blockIdx.x * 128;
  const int wm = (wid & 3) * 32, wn = (wid >> 2) * 64;
  const int lrow = lane & 15, lhalf = lane >> 4;
  const int rrow = t & 127;
  const int koff = (t >> 7) * 32;            // this thread stages 32 k-elements

  const bf16_t* gaBase = A  + (size_t)(m0 + rrow) * K + koff;
  const bf16_t* gbBase = Bw + (size_t)(n0 + rrow) * K + koff;

  v8f acc[2][4] = {};
  bf16x8 pfa[4], pfb[4];

  // prologue: stage K-tile 0 into registers
  #pragma unroll
  for (int u = 0; u < 4; u++) {
    pfa[u] = *(const bf16x8*)(gaBase + u * 8);
    pfb[u] = *(const bf16x8*)(gbBase + u * 8);
  }

  const int nT = K >> 6;                     // K / 64
  for (int kt = 0; kt < nT; kt++) {
    __syncthreads();                         // previous compute done
    #pragma unroll
    for (int u = 0; u < 4; u++) {
      *(bf16x8*)&As[rrow][koff + u * 8] = pfa[u];
      *(bf16x8*)&Bs[rrow][koff + u * 8] = pfb[u];
    }
    __syncthreads();

    if (kt + 1 < nT) {                       // prefetch next tile during compute
      const bf16_t* ga = gaBase + (kt + 1) * 64;
      const bf16_t* gb = gbBase + (kt + 1) * 64;
      #pragma unroll
      for (int u = 0; u < 4; u++) {
        pfa[u] = *(const bf16x8*)(ga + u * 8);
        pfb[u] = *(const bf16x8*)(gb + u * 8);
      }
    }

    #pragma unroll
    for (int ks = 0; ks < 64; ks += 32) {
      FragA af[2], bfr[4];
      #pragma unroll
      for (int i = 0; i < 2; i++) {          // A frag: lanes0-15 K{0..7,16..23}
        const int mr = wm + i * 16 + lrow;
        const int kb = ks + lhalf * 8;
        af[i].h[0] = *(const bf16x8*)&As[mr][kb];
        af[i].h[1] = *(const bf16x8*)&As[mr][kb + 16];
      }
      #pragma unroll
      for (int j = 0; j < 4; j++) {          // B frag: lanes0-15 K{0..7,16..23}
        const int nr = wn + j * 16 + lrow;
        const int kk = ks + lhalf * 8;
        bfr[j].h[0] = *(const bf16x8*)&Bs[nr][kk];
        bfr[j].h[1] = *(const bf16x8*)&Bs[nr][kk + 16];
      }
      #pragma unroll
      for (int i = 0; i < 2; i++)
        #pragma unroll
        for (int j = 0; j < 4; j++)
          acc[i][j] = wmma_bf16(af[i].v, bfr[j].v, acc[i][j]);
    }
  }

  #pragma unroll
  for (int i = 0; i < 2; i++) {
    #pragma unroll
    for (int j = 0; j < 4; j++) {
      const int col = n0 + wn + j * 16 + lrow;
      float bv = 0.0f;
      if (WRITE_F32 && bias) bv = bias[col];
      #pragma unroll
      for (int v = 0; v < 8; v++) {
        const int row = m0 + wm + i * 16 + v + 8 * lhalf;   // C layout
        float val = acc[i][j][v];
        if (WRITE_F32) {
          Cf[(size_t)row * N + col] = val + bv;
        } else {
          if (QKV_SCALE && col < scaleCols) val *= scale;
          Cb[(size_t)row * N + col] = f2bf(val);
        }
      }
    }
  }
}

// ---------------------------------------------------------------- attention
// qkv: bf16 [B*N][3072] (q pre-scaled by 1/8). out: bf16 [B*N][1024].
// Block = 8 waves; wave owns 16 query rows; keys streamed in 64-blocks with
// register prefetch of the next K/V block during compute.
__global__ __launch_bounds__(256)
void attn_k(const bf16_t* __restrict__ qkv, bf16_t* __restrict__ outb)
{
  __shared__ bf16_t Kls[64][72];        // [key][d]  -> B frag of Q@K^T
  __shared__ bf16_t Vt [64][72];        // [d][key]  -> B frag of P@V
  __shared__ bf16_t Pls[8][16][72];     // per-wave P scratch (C->A layout)

  const int bid  = blockIdx.x;
  const int qblk = bid & 15;            // N/128 = 16
  const int h    = (bid >> 4) & 15;
  const int b    = bid >> 8;
  const int t    = threadIdx.x;
  const int lane = t & 31, wid = t >> 5;
  const int lrow = lane & 15, lhalf = lane >> 4;
  const size_t rowBase = (size_t)b * 2048;
  const int QKVN = 3072;

  // Q fragments: 16 rows x 64 d -> two 16x32 A frags, registers for the run
  FragA qa[2];
  {
    const int qrow = qblk * 128 + wid * 16 + lrow;
    const bf16_t* qp = qkv + (rowBase + qrow) * QKVN + h * 64;
    const int kb = lhalf * 8;
    #pragma unroll
    for (int f = 0; f < 2; f++) {
      qa[f].h[0] = *(const bf16x8*)(qp + f * 32 + kb);
      qa[f].h[1] = *(const bf16x8*)(qp + f * 32 + kb + 16);
    }
  }

  float m_r[8], l_r[8];
  #pragma unroll
  for (int v = 0; v < 8; v++) { m_r[v] = -3.0e38f; l_r[v] = 0.0f; }
  v8f o[4] = {};

  const int r   = t & 63;               // key row staged by this thread
  const int seg = t >> 6;               // 16-wide d segment

  const bf16_t* kpB = qkv + (rowBase + r) * QKVN + 1024 + h * 64 + seg * 16;
  bf16x8 pk0 = *(const bf16x8*)kpB;
  bf16x8 pk1 = *(const bf16x8*)(kpB + 8);
  bf16x8 pv0 = *(const bf16x8*)(kpB + 1024);
  bf16x8 pv1 = *(const bf16x8*)(kpB + 1032);

  for (int kb0 = 0; kb0 < 2048; kb0 += 64) {
    __syncthreads();                    // previous block's compute done
    *(bf16x8*)&Kls[r][seg * 16]     = pk0;
    *(bf16x8*)&Kls[r][seg * 16 + 8] = pk1;
    #pragma unroll
    for (int i = 0; i < 8; i++) Vt[seg * 16 + i][r]     = pv0[i];
    #pragma unroll
    for (int i = 0; i < 8; i++) Vt[seg * 16 + 8 + i][r] = pv1[i];
    __syncthreads();

    if (kb0 + 64 < 2048) {              // prefetch next K/V block
      const bf16_t* kp = kpB + (size_t)(kb0 + 64) * QKVN;
      pk0 = *(const bf16x8*)kp;
      pk1 = *(const bf16x8*)(kp + 8);
      pv0 = *(const bf16x8*)(kp + 1024);
      pv1 = *(const bf16x8*)(kp + 1032);
    }

    // S = Q @ K^T   (16 x 64, f32 accum)
    v8f s[4];
    #pragma unroll
    for (int j = 0; j < 4; j++) {
      v8f sj = {};
      const int nr = j * 16 + lrow;
      const int kk = lhalf * 8;
      #pragma unroll
      for (int f = 0; f < 2; f++) {
        FragA bk;
        bk.h[0] = *(const bf16x8*)&Kls[nr][f * 32 + kk];
        bk.h[1] = *(const bf16x8*)&Kls[nr][f * 32 + kk + 16];
        sj = wmma_bf16(qa[f].v, bk.v, sj);
      }
      s[j] = sj;
    }

    // online softmax (row = v + 8*lhalf; a row lives in one 16-lane half)
    #pragma unroll
    for (int v = 0; v < 8; v++) {
      float mx = fmaxf(fmaxf(s[0][v], s[1][v]), fmaxf(s[2][v], s[3][v]));
      #pragma unroll
      for (int off = 8; off >= 1; off >>= 1)
        mx = fmaxf(mx, __shfl_xor(mx, off, 32));
      const float mnew  = fmaxf(m_r[v], mx);
      const float alpha = __expf(m_r[v] - mnew);
      m_r[v] = mnew;
      float rs = 0.0f;
      #pragma unroll
      for (int j = 0; j < 4; j++) {
        const float p = __expf(s[j][v] - mnew);
        s[j][v] = p;
        rs += p;
      }
      #pragma unroll
      for (int off = 8; off >= 1; off >>= 1)
        rs += __shfl_xor(rs, off, 32);
      l_r[v] = l_r[v] * alpha + rs;
      #pragma unroll
      for (int j = 0; j < 4; j++) o[j][v] *= alpha;
    }

    // P (C layout, f32) -> bf16 LDS -> A fragments (wave-local round trip)
    #pragma unroll
    for (int j = 0; j < 4; j++)
      #pragma unroll
      for (int v = 0; v < 8; v++)
        Pls[wid][v + 8 * lhalf][j * 16 + lrow] = f2bf(s[j][v]);
    __builtin_amdgcn_wave_barrier();

    FragA pa[2];
    {
      const int kb = lhalf * 8;
      #pragma unroll
      for (int f = 0; f < 2; f++) {
        pa[f].h[0] = *(const bf16x8*)&Pls[wid][lrow][f * 32 + kb];
        pa[f].h[1] = *(const bf16x8*)&Pls[wid][lrow][f * 32 + kb + 16];
      }
    }

    // O += P @ V
    #pragma unroll
    for (int jd = 0; jd < 4; jd++) {
      const int dn = jd * 16 + lrow;
      const int kk = lhalf * 8;
      v8f oj = o[jd];
      #pragma unroll
      for (int f = 0; f < 2; f++) {
        FragA bv;
        bv.h[0] = *(const bf16x8*)&Vt[dn][f * 32 + kk];
        bv.h[1] = *(const bf16x8*)&Vt[dn][f * 32 + kk + 16];
        oj = wmma_bf16(pa[f].v, bv.v, oj);
      }
      o[jd] = oj;
    }
  }

  // normalize, store bf16 into (B*N, 1024) for the projection GEMM
  #pragma unroll
  for (int v = 0; v < 8; v++) {
    const float inv = 1.0f / l_r[v];
    const size_t row = rowBase + qblk * 128 + wid * 16 + v + 8 * lhalf;
    #pragma unroll
    for (int jd = 0; jd < 4; jd++)
      outb[row * 1024 + h * 64 + jd * 16 + lrow] = f2bf(o[jd][v] * inv);
  }
}

// ---------------------------------------------------------------- launcher
extern "C" void kernel_launch(void* const* d_in, const int* in_sizes, int n_in,
                              void* d_out, int out_size, void* d_ws, size_t ws_size,
                              hipStream_t stream)
{
  const float* x      = (const float*)d_in[0];   // (4,2048,1024)
  const float* w_qkv  = (const float*)d_in[1];   // (3072,1024)
  const float* w_proj = (const float*)d_in[2];   // (1024,1024)
  const float* b_proj = (const float*)d_in[3];   // (1024,)
  float* out = (float*)d_out;                    // (4,2048,1024) f32

  char* ws = (char*)d_ws;
  bf16_t* xb     = (bf16_t*)(ws);                 // 16 MB
  bf16_t* wqkvb  = (bf16_t*)(ws + 16777216);      //  6 MB
  bf16_t* wprojb = (bf16_t*)(ws + 23068672);      //  2 MB
  bf16_t* qkvb   = (bf16_t*)(ws + 25165824);      // 48 MB (B*N x 3072 bf16)
  bf16_t* attnb  = (bf16_t*)(ws + 75497472);      // 16 MB (B*N x 1024 bf16)

  cvt_k<<<1024, 256, 0, stream>>>(x,      xb,     (8192 * 1024) / 4);
  cvt_k<<<512,  256, 0, stream>>>(w_qkv,  wqkvb,  (3072 * 1024) / 4);
  cvt_k<<<256,  256, 0, stream>>>(w_proj, wprojb, (1024 * 1024) / 4);

  // qkv = x @ w_qkv^T, q-third scaled by 1/sqrt(64)
  gemm_bf16_k<false, true><<<dim3(24, 64), 256, 0, stream>>>(
      xb, wqkvb, qkvb, nullptr, nullptr, 8192, 3072, 1024, 0.125f, 1024);

  // flash attention per (b, h, 128-query block)
  attn_k<<<1024, 256, 0, stream>>>(qkvb, attnb);

  // out = attn @ w_proj^T + b_proj
  gemm_bf16_k<true, false><<<dim3(8, 64), 256, 0, stream>>>(
      attnb, wprojb, nullptr, out, b_proj, 8192, 1024, 1024, 1.0f, 0);
}
